// DualGuidanceAttentionFusion_62268435858085
// MI455X (gfx1250) — compile-verified, matrix operand order checked by reading, not run on hardware
//
#include <hip/hip_runtime.h>
#include <math.h>

// ---------- shapes ----------
#define MODEL_DIM 128
#define HEADS 4
#define PTS 4
#define HH 48
#define WW 48
#define NPIX (HH*WW)        // 2304
#define NC 3
#define NG 3                // granularities
#define PARAM_OUT (HEADS*PTS*6) // 96

typedef _Float16 h16;
typedef __attribute__((ext_vector_type(8)))  _Float16 v8h;
typedef __attribute__((ext_vector_type(16))) _Float16 v16h;
typedef __attribute__((ext_vector_type(8)))  float    v8f;

// ---------- workspace layout (bytes, 256-aligned) ----------
#define OFF_WVAL  0u          //  32768  value_w f16 (128x128)
#define OFF_WOUT  32768u      //  32768  out_w   f16 (128x128)
#define OFF_WPAR  65536u      //  24576  param_w f16 (96x128)
#define OFF_QH    90112u      // 589824  query f16 [pix][128]
#define OFF_PAR   679936u     // 884736  params f32 [pix][96]
#define OFF_LOC   1564672u    // 294912  loc f32 [h][pix][pt][2]
#define OFF_SW    1859584u    // 147456  spatial w f32 [h][pix][pt]
#define OFF_GW    2007040u    // 442368  gran w f32 [h][pix][pt][3]
#define OFF_ACT   2449408u    // 5308416 act f16 [gn][pix][128]
#define OFF_VAL   7757824u    // 10616832 val f32 [gn][pix][128]
#define OFF_OPRE  18374656u   // 589824  pre-out f16 [pix][128]

// Build a 16x32 f16 WMMA fragment per ISA layout: two contiguous 8-half K
// chunks at p and p+16 (caller applies the lane's K base).
__device__ __forceinline__ v16h make_frag(const h16* p) {
    v8h lo = *(const v8h*)p;
    v8h hi = *(const v8h*)(p + 16);
    return __builtin_shufflevector(lo, hi, 0,1,2,3,4,5,6,7,8,9,10,11,12,13,14,15);
}

// ---------- f32 -> f16 convert ----------
__global__ void cvt_f16_kernel(const float* __restrict__ s, h16* __restrict__ d, int n) {
    int i = blockIdx.x * blockDim.x + threadIdx.x;
    if (i < n) d[i] = (h16)s[i];
}

// ---------- collab features NCHW f32 -> [gn][pix][c] f16 (LDS tile transpose) ----
// Coalesced f32 reads (lanes sweep pixels), padded LDS (stride 130 halfs, no
// bank conflicts), coalesced f16 writes (lanes sweep channels).
__global__ void cvt_acts_kernel(const float* __restrict__ g0, const float* __restrict__ g1,
                                const float* __restrict__ g2, h16* __restrict__ act) {
    __shared__ h16 tile[64 * 130];
    int gn = blockIdx.y;            // 0..8
    int pix0 = blockIdx.x * 64;     // 36 tiles of 64 pixels
    int g = gn / NC, n = gn % NC;
    const float* src = (g == 0) ? g0 : ((g == 1) ? g1 : g2);
    int t = threadIdx.x;            // 256 threads
#pragma unroll
    for (int i = 0; i < 32; ++i) {
        int e = t + 256 * i;        // 64px x 128ch = 8192 elements
        int pixl = e & 63;
        int c = e >> 6;
        tile[pixl * 130 + c] = (h16)src[(size_t)(n * MODEL_DIM + c) * NPIX + pix0 + pixl];
    }
    __syncthreads();
#pragma unroll
    for (int i = 0; i < 32; ++i) {
        int e = t + 256 * i;
        int c = e & 127;
        int pixl = e >> 7;
        act[((size_t)gn * NPIX + pix0 + pixl) * MODEL_DIM + c] = tile[pixl * 130 + c];
    }
}

// ---------- demand MLP + query (f16, [pix][c]) ----------
__global__ void query_kernel(const float* __restrict__ ego, const float* __restrict__ dem,
                             const float* __restrict__ pos,
                             const float* __restrict__ w1, const float* __restrict__ b1,
                             const float* __restrict__ w2, const float* __restrict__ b2,
                             h16* __restrict__ qh) {
    __shared__ float hid[64];
    int pix = blockIdx.x;
    int t = threadIdx.x;
    float d0 = dem[0 * NPIX + pix], d1 = dem[1 * NPIX + pix], d2 = dem[2 * NPIX + pix];
    if (t < 64) {
        float s = w1[t * 3 + 0] * d0 + w1[t * 3 + 1] * d1 + w1[t * 3 + 2] * d2 + b1[t];
        hid[t] = s > 0.f ? s : 0.f;
    }
    __syncthreads();
    float e = b2[t];
#pragma unroll 8
    for (int j = 0; j < 64; ++j) e += w2[t * 64 + j] * hid[j];
    float q = ego[t * NPIX + pix] + e + pos[t * NPIX + pix];
    qh[pix * MODEL_DIM + t] = (h16)q;
}

// ---------- WMMA GEMM: params[pix][96] = param_w(96x128) x query^T + b ----------
__global__ void param_gemm_kernel(const h16* __restrict__ W, const h16* __restrict__ B,
                                  const float* __restrict__ bias, float* __restrict__ out) {
    int lane = threadIdx.x;
    int l16 = lane & 15;
    int hi8 = (lane >> 4) & 1;
    const h16* Arow = W + (blockIdx.y * 16 + l16) * MODEL_DIM;
    const h16* Brow = B + (size_t)(blockIdx.x * 16 + l16) * MODEL_DIM;
    v8f acc = {0.f,0.f,0.f,0.f,0.f,0.f,0.f,0.f};
#pragma unroll
    for (int ks = 0; ks < 4; ++ks) {
        int kb = ks * 32 + hi8 * 8;
        v16h a = make_frag(Arow + kb);
        v16h b = make_frag(Brow + kb);
        acc = __builtin_amdgcn_wmma_f32_16x16x32_f16(false, a, false, b, (short)0, acc, false, false);
    }
    union { v8f v; float f[8]; } u; u.v = acc;
    int oc0 = blockIdx.y * 16 + hi8 * 8;
    float* op = out + (size_t)(blockIdx.x * 16 + l16) * PARAM_OUT + oc0;
#pragma unroll
    for (int r = 0; r < 8; ++r) op[r] = u.f[r] + bias[oc0 + r];
}

// ---------- per (head,pix) parameter post-processing ----------
__global__ void parampost_kernel(const float* __restrict__ params, float* __restrict__ loc,
                                 float* __restrict__ sw, float* __restrict__ gw) {
    int id = blockIdx.x * blockDim.x + threadIdx.x;
    if (id >= HEADS * NPIX) return;
    int h = id / NPIX, pix = id % NPIX;
    const float* p = params + (size_t)pix * PARAM_OUT + h * (PTS * 6);
    int hy = pix / WW, wx = pix % WW;
    float refx = -1.f + 2.f * (float)wx / 47.f;
    float refy = -1.f + 2.f * (float)hy / 47.f;
    size_t base = (size_t)h * NPIX + pix;
#pragma unroll
    for (int pt = 0; pt < PTS; ++pt) {
        loc[(base * PTS + pt) * 2 + 0] = refx + tanhf(p[pt * 2 + 0]);
        loc[(base * PTS + pt) * 2 + 1] = refy + tanhf(p[pt * 2 + 1]);
    }
    // spatial softmax over PTS
    {
        float m = p[8];
        for (int pt = 1; pt < PTS; ++pt) m = fmaxf(m, p[8 + pt]);
        float e[PTS], s = 0.f;
#pragma unroll
        for (int pt = 0; pt < PTS; ++pt) { e[pt] = expf(p[8 + pt] - m); s += e[pt]; }
        float inv = 1.f / s;
#pragma unroll
        for (int pt = 0; pt < PTS; ++pt) sw[base * PTS + pt] = e[pt] * inv;
    }
    // granularity softmax over 3, per point
#pragma unroll
    for (int pt = 0; pt < PTS; ++pt) {
        const float* q = p + 12 + pt * 3;
        float m = fmaxf(q[0], fmaxf(q[1], q[2]));
        float e0 = expf(q[0] - m), e1 = expf(q[1] - m), e2 = expf(q[2] - m);
        float inv = 1.f / (e0 + e1 + e2);
        gw[(base * PTS + pt) * 3 + 0] = e0 * inv;
        gw[(base * PTS + pt) * 3 + 1] = e1 * inv;
        gw[(base * PTS + pt) * 3 + 2] = e2 * inv;
    }
}

// ---------- WMMA GEMM: val[gn][pix][128] = value_w(128x128) x act^T + b ----------
// N-blocked: each wave computes 16 oc x 64 px (4 tiles), reusing each A
// fragment across 4 WMMAs. Prefetch pulls the next K chunk of B rows.
__global__ void value_gemm_kernel(const h16* __restrict__ W, const h16* __restrict__ act,
                                  const float* __restrict__ bias, float* __restrict__ val) {
    int lane = threadIdx.x;
    int l16 = lane & 15;
    int hi8 = (lane >> 4) & 1;
    size_t gnbase = (size_t)blockIdx.z * NPIX;
    int pix0 = blockIdx.x * 64;
    const h16* Arow = W + (blockIdx.y * 16 + l16) * MODEL_DIM;
    const h16* Brow = act + (gnbase + pix0 + l16) * MODEL_DIM;
    v8f acc[4];
#pragma unroll
    for (int j = 0; j < 4; ++j) acc[j] = (v8f){0.f,0.f,0.f,0.f,0.f,0.f,0.f,0.f};
#pragma unroll
    for (int ks = 0; ks < 4; ++ks) {
        int kb = ks * 32 + hi8 * 8;
        v16h a = make_frag(Arow + kb);
        if (ks < 3) {
            __builtin_prefetch(Brow + kb + 32, 0, 3);                    // next K chunk, tile 0
            __builtin_prefetch(Brow + 3 * 16 * MODEL_DIM + kb + 32, 0, 3); // next K chunk, tile 3
        }
#pragma unroll
        for (int j = 0; j < 4; ++j) {
            v16h b = make_frag(Brow + (size_t)j * 16 * MODEL_DIM + kb);
            acc[j] = __builtin_amdgcn_wmma_f32_16x16x32_f16(false, a, false, b, (short)0, acc[j], false, false);
        }
    }
    int oc0 = blockIdx.y * 16 + hi8 * 8;
    float4 b0 = *(const float4*)(bias + oc0);
    float4 b1 = *(const float4*)(bias + oc0 + 4);
#pragma unroll
    for (int j = 0; j < 4; ++j) {
        union { v8f v; float f[8]; } u; u.v = acc[j];
        float* op = val + (gnbase + pix0 + j * 16 + l16) * MODEL_DIM + oc0;
        float4 o0 = {u.f[0] + b0.x, u.f[1] + b0.y, u.f[2] + b0.z, u.f[3] + b0.w};
        float4 o1 = {u.f[4] + b1.x, u.f[5] + b1.y, u.f[6] + b1.z, u.f[7] + b1.w};
        *(float4*)op = o0;
        *(float4*)(op + 4) = o1;
    }
}

// ---------- fused bilinear sample + gran mix + mask-normalized collab avg ----------
__global__ void sample_kernel(const float* __restrict__ val, const int* __restrict__ masks,
                              const float* __restrict__ loc, const float* __restrict__ sw,
                              const float* __restrict__ gw, h16* __restrict__ opre) {
    __shared__ int   s_idx[PTS][4];
    __shared__ float s_w[PTS][4];
    __shared__ float s_sm[PTS][NC];
    int pix = blockIdx.x;
    int c = threadIdx.x;
    float outacc = 0.f;
    for (int h = 0; h < HEADS; ++h) {
        size_t base = (size_t)h * NPIX + pix;
        if (c < PTS) {
            int pt = c;
            float lx = loc[(base * PTS + pt) * 2 + 0];
            float ly = loc[(base * PTS + pt) * 2 + 1];
            float x = (lx + 1.f) * (WW * 0.5f) - 0.5f;
            float y = (ly + 1.f) * (HH * 0.5f) - 0.5f;
            float x0f = floorf(x), y0f = floorf(y);
            int x0 = (int)x0f, y0 = (int)y0f, x1 = x0 + 1, y1 = y0 + 1;
            float wxf = x - x0f, wyf = y - y0f;
            int cx0 = min(max(x0, 0), WW - 1), cx1 = min(max(x1, 0), WW - 1);
            int cy0 = min(max(y0, 0), HH - 1), cy1 = min(max(y1, 0), HH - 1);
            float v00 = (x0 >= 0 && x0 < WW && y0 >= 0 && y0 < HH) ? 1.f : 0.f;
            float v01 = (x1 >= 0 && x1 < WW && y0 >= 0 && y0 < HH) ? 1.f : 0.f;
            float v10 = (x0 >= 0 && x0 < WW && y1 >= 0 && y1 < HH) ? 1.f : 0.f;
            float v11 = (x1 >= 0 && x1 < WW && y1 >= 0 && y1 < HH) ? 1.f : 0.f;
            s_idx[pt][0] = cy0 * WW + cx0; s_w[pt][0] = (1.f - wxf) * (1.f - wyf) * v00;
            s_idx[pt][1] = cy0 * WW + cx1; s_w[pt][1] = wxf * (1.f - wyf) * v01;
            s_idx[pt][2] = cy1 * WW + cx0; s_w[pt][2] = (1.f - wxf) * wyf * v10;
            s_idx[pt][3] = cy1 * WW + cx1; s_w[pt][3] = wxf * wyf * v11;
        }
        if (c < PTS * NC) {
            int pt = c / NC, n = c % NC;
            float lx = loc[(base * PTS + pt) * 2 + 0];
            float ly = loc[(base * PTS + pt) * 2 + 1];
            float x = (lx + 1.f) * (WW * 0.5f) - 0.5f;
            float y = (ly + 1.f) * (HH * 0.5f) - 0.5f;
            int xi = (int)nearbyintf(x), yi = (int)nearbyintf(y);
            bool valid = (xi >= 0) && (xi < WW) && (yi >= 0) && (yi < HH);
            int idx = min(max(yi, 0), HH - 1) * WW + min(max(xi, 0), WW - 1);
            int um = masks[(n * 3 + 0) * NPIX + idx] + masks[(n * 3 + 1) * NPIX + idx] +
                     masks[(n * 3 + 2) * NPIX + idx];
            s_sm[pt][n] = (valid && (um > 0)) ? 1.f : 0.f;
        }
        __syncthreads();
        const float* gwp = gw + base * PTS * 3;
        const float* swp = sw + base * PTS;
#pragma unroll
        for (int pt = 0; pt < PTS; ++pt) {
            float num = s_sm[pt][0] + s_sm[pt][1] + s_sm[pt][2];
            if (num > 0.f) {
                float g0 = gwp[pt * 3 + 0], g1 = gwp[pt * 3 + 1], g2 = gwp[pt * 3 + 2];
                float w0 = s_w[pt][0], w1 = s_w[pt][1], w2 = s_w[pt][2], w3 = s_w[pt][3];
                int i0 = s_idx[pt][0], i1 = s_idx[pt][1], i2 = s_idx[pt][2], i3 = s_idx[pt][3];
                float ssum = 0.f;
#pragma unroll
                for (int n = 0; n < NC; ++n) {
                    if (s_sm[pt][n] > 0.f) {
                        float b = 0.f;
#pragma unroll
                        for (int g = 0; g < NG; ++g) {
                            const float* vb = val + (size_t)((g * NC + n) * NPIX) * MODEL_DIM + c;
                            float bv = w0 * vb[(size_t)i0 * MODEL_DIM] + w1 * vb[(size_t)i1 * MODEL_DIM] +
                                       w2 * vb[(size_t)i2 * MODEL_DIM] + w3 * vb[(size_t)i3 * MODEL_DIM];
                            b += ((g == 0) ? g0 : (g == 1) ? g1 : g2) * bv;
                        }
                        ssum += b;
                    }
                }
                outacc += (ssum / fmaxf(num, 1e-6f)) * swp[pt];
            }
        }
        __syncthreads();
    }
    opre[(size_t)pix * MODEL_DIM + c] = (h16)outacc;
}

// ---------- WMMA GEMM: d_out NCHW = out_w(128x128) x opre^T + b ----------
__global__ void out_gemm_kernel(const h16* __restrict__ W, const h16* __restrict__ B,
                                const float* __restrict__ bias, float* __restrict__ out) {
    int lane = threadIdx.x;
    int l16 = lane & 15;
    int hi8 = (lane >> 4) & 1;
    const h16* Arow = W + (blockIdx.y * 16 + l16) * MODEL_DIM;
    const h16* Brow = B + (size_t)(blockIdx.x * 16 + l16) * MODEL_DIM;
    v8f acc = {0.f,0.f,0.f,0.f,0.f,0.f,0.f,0.f};
#pragma unroll
    for (int ks = 0; ks < 4; ++ks) {
        int kb = ks * 32 + hi8 * 8;
        v16h a = make_frag(Arow + kb);
        v16h b = make_frag(Brow + kb);
        acc = __builtin_amdgcn_wmma_f32_16x16x32_f16(false, a, false, b, (short)0, acc, false, false);
    }
    union { v8f v; float f[8]; } u; u.v = acc;
    int n = blockIdx.x * 16 + l16;
#pragma unroll
    for (int r = 0; r < 8; ++r) {
        int oc = blockIdx.y * 16 + hi8 * 8 + r;
        out[(size_t)oc * NPIX + n] = u.f[r] + bias[oc];
    }
}

extern "C" void kernel_launch(void* const* d_in, const int* in_sizes, int n_in,
                              void* d_out, int out_size, void* d_ws, size_t ws_size,
                              hipStream_t stream) {
    const float* ego   = (const float*)d_in[0];
    const float* dem   = (const float*)d_in[1];
    const float* g0    = (const float*)d_in[2];
    const float* g1    = (const float*)d_in[3];
    const float* g2    = (const float*)d_in[4];
    const int*   masks = (const int*)d_in[5];
    const float* pos   = (const float*)d_in[6];
    const float* dw1   = (const float*)d_in[7];
    const float* db1   = (const float*)d_in[8];
    const float* dw2   = (const float*)d_in[9];
    const float* db2   = (const float*)d_in[10];
    const float* pw    = (const float*)d_in[11];
    const float* pb    = (const float*)d_in[12];
    const float* vw    = (const float*)d_in[13];
    const float* vb    = (const float*)d_in[14];
    const float* ow    = (const float*)d_in[15];
    const float* ob    = (const float*)d_in[16];
    float* outp = (float*)d_out;

    char* ws = (char*)d_ws;
    h16*   w_val_h = (h16*)(ws + OFF_WVAL);
    h16*   w_out_h = (h16*)(ws + OFF_WOUT);
    h16*   w_par_h = (h16*)(ws + OFF_WPAR);
    h16*   q_h     = (h16*)(ws + OFF_QH);
    float* params  = (float*)(ws + OFF_PAR);
    float* locb    = (float*)(ws + OFF_LOC);
    float* swb     = (float*)(ws + OFF_SW);
    float* gwb     = (float*)(ws + OFF_GW);
    h16*   act_h   = (h16*)(ws + OFF_ACT);
    float* val_f   = (float*)(ws + OFF_VAL);
    h16*   opre_h  = (h16*)(ws + OFF_OPRE);

    // 1. weight conversion
    cvt_f16_kernel<<<(MODEL_DIM*MODEL_DIM + 255)/256, 256, 0, stream>>>(vw, w_val_h, MODEL_DIM*MODEL_DIM);
    cvt_f16_kernel<<<(MODEL_DIM*MODEL_DIM + 255)/256, 256, 0, stream>>>(ow, w_out_h, MODEL_DIM*MODEL_DIM);
    cvt_f16_kernel<<<(PARAM_OUT*MODEL_DIM + 255)/256, 256, 0, stream>>>(pw, w_par_h, PARAM_OUT*MODEL_DIM);

    // 2. demand MLP + query
    query_kernel<<<NPIX, MODEL_DIM, 0, stream>>>(ego, dem, pos, dw1, db1, dw2, db2, q_h);

    // 3. param projection (WMMA)
    param_gemm_kernel<<<dim3(NPIX/16, PARAM_OUT/16), 32, 0, stream>>>(w_par_h, q_h, pb, params);

    // 4. tanh / softmax post-processing
    parampost_kernel<<<(HEADS*NPIX + 255)/256, 256, 0, stream>>>(params, locb, swb, gwb);

    // 5. transpose+convert collab features (LDS tile transpose, coalesced both sides)
    cvt_acts_kernel<<<dim3(NPIX/64, NG*NC), 256, 0, stream>>>(g0, g1, g2, act_h);

    // 6. value projection (WMMA, 16oc x 64px per wave), 9 images
    value_gemm_kernel<<<dim3(NPIX/64, MODEL_DIM/16, NG*NC), 32, 0, stream>>>(w_val_h, act_h, vb, val_f);

    // 7. fused sampling / aggregation
    sample_kernel<<<NPIX, MODEL_DIM, 0, stream>>>(val_f, masks, locb, swb, gwb, opre_h);

    // 8. output projection (WMMA) -> NCHW f32
    out_gemm_kernel<<<dim3(NPIX/16, MODEL_DIM/16), 32, 0, stream>>>(w_out_h, opre_h, ob, outp);
}
